// PillarVoxelTower_48086453846504
// MI455X (gfx1250) — matrix operand order repaired and checked by reading, not compiled, not run on hardware
//
#include <hip/hip_runtime.h>

typedef __attribute__((ext_vector_type(16))) _Float16 v16h;
typedef __attribute__((ext_vector_type(8)))  _Float16 v8h;
typedef __attribute__((ext_vector_type(8)))  float    v8f;

// ---------------- problem constants ----------------
#define BB     8
#define GH     40
#define GW     40
#define NPTS   32
#define CF     256
#define M_PIL  (BB*GH*GW)          // 12800 pillars
#define O1     1024
#define K1     (256*9)             // 2304
#define M1     (BB*20*20)          // 3200 conv1 output positions
#define O2     4096
#define K2     1024
#define M2     (BB*10*10)          // 800 conv2 output positions
#define OUT_MAIN (M2*O2)           // 3276800
#define VXc    0.025f
#define VYc    0.025f
#define VZc    1.0f
#define X0c    (-0.5f)
#define Y0c    (-0.5f)
#define Z0c    0.0f

// ---------------- workspace layout (bytes, 256B aligned) ----------------
#define WS_CANVAS 0u               // f16 [8][40][40][256]      = 6,553,600
#define WS_W1T    6553600u         // f16 [2304][1024]          = 4,718,592
#define WS_ACT1   11272192u        // f16 [3200][1024]          = 6,553,600
#define WS_W2T    17825792u        // f16 [1024][4096]          = 8,388,608
#define WS_X2     26214400u        // f32 [800][4096]           = 13,107,200
#define WS_OFFS   39321600u        // int [8]
#define WS_ZERO   39321856u        // 256B zero page for OOB im2col taps

// LDS byte offset of a __shared__ object = low 32 bits of its generic address
// (ISA 10.2: LDS aperture)
__device__ __forceinline__ unsigned lds_off(const void* p) {
    return (unsigned)(size_t)p;
}
__device__ __forceinline__ void async_copy_b128(unsigned lds_dst, const void* gsrc) {
    asm volatile("global_load_async_to_lds_b128 %0, %1, off"
                 :: "v"(lds_dst), "v"(gsrc) : "memory");
}
__device__ __forceinline__ void async_copy_b64(unsigned lds_dst, const void* gsrc) {
    asm volatile("global_load_async_to_lds_b64 %0, %1, off"
                 :: "v"(lds_dst), "v"(gsrc) : "memory");
}
__device__ __forceinline__ void wait_async0() {
    asm volatile("s_wait_asynccnt 0x0" ::: "memory");
}

// ===================================================================
// Weight repack: conv1_w (O,C,3,3) f32 -> w1t[(tap*256+c)][o] f16
// Also zeroes valid counters and the OOB zero page (every call).
// ===================================================================
__global__ __launch_bounds__(256) void k_prep_w1(const float* __restrict__ w,
                                                 _Float16* __restrict__ w1t,
                                                 int* __restrict__ offs,
                                                 float* __restrict__ zerop) {
    int idx = blockIdx.x * 256 + threadIdx.x;
    if (idx < 8)  offs[idx] = 0;
    if (idx < 64) zerop[idx] = 0.0f;
    if (idx >= O1 * K1) return;
    int o = idx / K1;
    int r = idx - o * K1;
    int c = r / 9;
    int t = r - c * 9;                       // ky*3+kx
    w1t[(t * 256 + c) * O1 + o] = (_Float16)w[idx];
}

// conv2_w (O,C) f32 -> w2t[c][o] f16
__global__ __launch_bounds__(256) void k_prep_w2(const float* __restrict__ w,
                                                 _Float16* __restrict__ w2t) {
    int idx = blockIdx.x * 256 + threadIdx.x;
    if (idx >= O2 * K2) return;
    int o = idx >> 10;
    int c = idx & 1023;
    w2t[c * O2 + o] = (_Float16)w[idx];
}

// ===================================================================
// PFN: per-pillar feature build + [32x12]x[12x256] GEMM + BN/ReLU/max
// ===================================================================
__global__ __launch_bounds__(256) void k_pfn(const float* __restrict__ voxels,
                                             const int*   __restrict__ num_points,
                                             const int*   __restrict__ coors,
                                             const float* __restrict__ w_pfn,
                                             const float* __restrict__ bn_g,
                                             const float* __restrict__ bn_b,
                                             const float* __restrict__ bn_m,
                                             const float* __restrict__ bn_v,
                                             _Float16*    __restrict__ canvas) {
    const int m   = blockIdx.x;
    const int tid = threadIdx.x;
    __shared__ __align__(16) float feats[NPTS][12];
    __shared__ float s_mean[3];

    const int np = num_points[m];
    const float inv_np = 1.0f / (float)(np > 1 ? np : 1);

    if (tid < NPTS) {
        const float mask = (tid < np) ? 1.0f : 0.0f;
        const float* vp = voxels + ((size_t)m * NPTS + tid) * 6;
        #pragma unroll
        for (int d = 0; d < 6; ++d) feats[tid][d] = vp[d] * mask;
    }
    __syncthreads();
    if (tid < 3) {
        float s = 0.f;
        #pragma unroll
        for (int n = 0; n < NPTS; ++n) s += feats[n][tid];
        s_mean[tid] = s * inv_np;
    }
    __syncthreads();
    if (tid < NPTS) {
        const float mask = (tid < np) ? 1.0f : 0.0f;
        const float cx = (float)coors[m * 4 + 3] * VXc + (VXc * 0.5f + X0c);
        const float cy = (float)coors[m * 4 + 2] * VYc + (VYc * 0.5f + Y0c);
        const float cz = (float)coors[m * 4 + 1] * VZc + (VZc * 0.5f + Z0c);
        const float ctr[3] = {cx, cy, cz};
        #pragma unroll
        for (int d = 0; d < 3; ++d) {
            const float xm = feats[tid][d];                // masked xyz
            feats[tid][6 + d] = (xm - s_mean[d]) * mask;
            feats[tid][9 + d] = (xm - ctr[d]) * mask;
        }
    }
    __syncthreads();

    const int f = tid;
    float wc[12];
    #pragma unroll
    for (int d = 0; d < 12; ++d) wc[d] = w_pfn[d * CF + f];
    const float scale = bn_g[f] * rsqrtf(bn_v[f] + 1e-3f);
    const float mu    = bn_m[f];
    const float beta  = bn_b[f];
    float mx = -3.4e38f;
    #pragma unroll 4
    for (int n = 0; n < NPTS; ++n) {
        float h = 0.f;
        #pragma unroll
        for (int d = 0; d < 12; ++d) h = fmaf(feats[n][d], wc[d], h);
        h = (h - mu) * scale + beta;
        h = h > 0.f ? h : 0.f;
        mx = mx > h ? mx : h;
    }
    const int b = coors[m * 4 + 0];
    const int y = coors[m * 4 + 2];
    const int x = coors[m * 4 + 3];
    canvas[((((b * GH) + y) * GW + x) << 8) + f] = (_Float16)mx;
}

// ===================================================================
// conv1 (3x3 s2 p1) implicit GEMM [3200 x 2304]x[2304 x 1024]
// 64x64 macro tile, 8 waves x 2 WMMA tiles, K-step 32,
// double-buffered LDS fed by global_load_async_to_lds_b128.
// ===================================================================
__global__ __launch_bounds__(256) void k_conv1(const _Float16* __restrict__ canvas,
                                               const _Float16* __restrict__ w1t,
                                               const float*    __restrict__ bias,
                                               const _Float16* __restrict__ zerop,
                                               _Float16*       __restrict__ act1) {
    __shared__ __align__(32) _Float16 sA[2][64 * 32];
    __shared__ __align__(32) _Float16 sB[2][32 * 64];
    const int tid  = threadIdx.x;
    const int lane = tid & 31;
    const int wave = tid >> 5;
    const int wm   = wave & 3;            // 4 M-tiles
    const int wnb  = (wave >> 2) * 2;     // 2 N-tiles per wave
    const int m0   = blockIdx.x * 64;
    const int n0   = blockIdx.y * 64;

    // A staging: 64 rows x 32 halfs, 16B/thread
    const int ar   = tid >> 2;
    const int aseg = (tid & 3) * 8;
    const int am   = m0 + ar;
    const int ab   = am / 400;
    const int arem = am - ab * 400;
    const int aoy  = arem / 20;
    const int aox  = arem - aoy * 20;
    // B staging: 32 rows x 64 halfs, 16B/thread
    const int bkr  = tid >> 3;
    const int bseg = (tid & 7) * 8;

    const unsigned dstA0 = lds_off(&sA[0][ar * 32 + aseg]);
    const unsigned dstA1 = lds_off(&sA[1][ar * 32 + aseg]);
    const unsigned dstB0 = lds_off(&sB[0][bkr * 64 + bseg]);
    const unsigned dstB1 = lds_off(&sB[1][bkr * 64 + bseg]);

    const int mrow  = wm * 16 + (lane & 15);
    const int khalf = (lane >> 4) * 8;

    v8f acc0 = {};
    v8f acc1 = {};

    #define C1_ISSUE(step)                                                        \
    {                                                                             \
        const int kb  = (step) * 32;                                              \
        const int tap = kb >> 8;                                                  \
        const int ky  = tap / 3;                                                  \
        const int kx  = tap - ky * 3;                                             \
        const int iy  = aoy * 2 - 1 + ky;                                         \
        const int ix  = aox * 2 - 1 + kx;                                         \
        const _Float16* ga = ((unsigned)iy < (unsigned)GH && (unsigned)ix < (unsigned)GW) \
            ? canvas + ((((ab * GH + iy) * GW + ix) << 8) + (kb & 255) + aseg)    \
            : zerop + aseg;                                                       \
        const _Float16* gb = w1t + (size_t)(kb + bkr) * O1 + n0 + bseg;           \
        async_copy_b128(((step) & 1) ? dstA1 : dstA0, ga);                        \
        async_copy_b128(((step) & 1) ? dstB1 : dstB0, gb);                        \
    }

    C1_ISSUE(0)
    for (int step = 0; step < K1 / 32; ++step) {
        wait_async0();               // own stage-(step) copies landed in LDS
        __syncthreads();             // everyone's landed; prior stage reads done
        if (step + 1 < K1 / 32) C1_ISSUE(step + 1)
        const int st = step & 1;

        const v8h alo = *(const v8h*)(&sA[st][mrow * 32 + khalf]);
        const v8h ahi = *(const v8h*)(&sA[st][mrow * 32 + 16 + khalf]);
        v16h a;
        #pragma unroll
        for (int i = 0; i < 8; ++i) { a[i] = alo[i]; a[i + 8] = ahi[i]; }
        v16h b0, b1;
        #pragma unroll
        for (int i = 0; i < 8; ++i) {
            b0[i]     = sB[st][(khalf + i) * 64 + (wnb + 0) * 16 + (lane & 15)];
            b0[i + 8] = sB[st][(16 + khalf + i) * 64 + (wnb + 0) * 16 + (lane & 15)];
            b1[i]     = sB[st][(khalf + i) * 64 + (wnb + 1) * 16 + (lane & 15)];
            b1[i + 8] = sB[st][(16 + khalf + i) * 64 + (wnb + 1) * 16 + (lane & 15)];
        }
        acc0 = __builtin_amdgcn_wmma_f32_16x16x32_f16(false, a, false, b0, (short)0, acc0, false, false);
        acc1 = __builtin_amdgcn_wmma_f32_16x16x32_f16(false, a, false, b1, (short)0, acc1, false, false);
    }
    #undef C1_ISSUE

    // epilogue: bias + ReLU, f16 NHWC activation for conv2
    const int mbase = m0 + wm * 16 + ((lane >> 4) ? 8 : 0);
    #pragma unroll
    for (int t = 0; t < 2; ++t) {
        const v8f acc = t ? acc1 : acc0;
        const int ng  = n0 + (wnb + t) * 16 + (lane & 15);
        const float bv = bias[ng];
        #pragma unroll
        for (int r = 0; r < 8; ++r) {
            float v = acc[r] + bv;
            v = v > 0.f ? v : 0.f;
            act1[(size_t)(mbase + r) * O1 + ng] = (_Float16)v;
        }
    }
}

// ===================================================================
// conv2 (1x1 s2) GEMM [800 x 1024]x[1024 x 4096]
// 32x128 macro tile, 8 waves x 2 WMMA tiles, async double buffering.
// ===================================================================
__global__ __launch_bounds__(256) void k_conv2(const _Float16* __restrict__ act1,
                                               const _Float16* __restrict__ w2t,
                                               const float*    __restrict__ bias,
                                               float*          __restrict__ x2) {
    __shared__ __align__(32) _Float16 sA[2][32 * 32];
    __shared__ __align__(32) _Float16 sB[2][32 * 128];
    const int tid  = threadIdx.x;
    const int lane = tid & 31;
    const int wave = tid >> 5;
    const int wm   = wave & 1;            // 2 M-tiles
    const int wnb  = (wave >> 1) * 2;     // 2 N-tiles per wave (8 total)
    const int m0   = blockIdx.x * 32;
    const int n0   = blockIdx.y * 128;

    // A staging: 32 rows x 32 halfs, 8B/thread (all 256 threads active)
    const int ar   = tid >> 3;
    const int aseg = (tid & 7) * 4;
    const int am   = m0 + ar;
    const int ab2  = am / 100;
    const int arem = am - ab2 * 100;
    const int oy2  = arem / 10;
    const int ox2  = arem - oy2 * 10;
    const int arow = ab2 * 400 + oy2 * 40 + ox2 * 2;   // act1 row at (2*oy2, 2*ox2)
    // B staging: 32 rows x 128 halfs, 32B/thread (2 x b128)
    const int bkr  = tid >> 3;
    const int bseg = (tid & 7) * 16;

    const unsigned dstA0  = lds_off(&sA[0][ar * 32 + aseg]);
    const unsigned dstA1  = lds_off(&sA[1][ar * 32 + aseg]);
    const unsigned dstB0a = lds_off(&sB[0][bkr * 128 + bseg]);
    const unsigned dstB0b = lds_off(&sB[0][bkr * 128 + bseg + 8]);
    const unsigned dstB1a = lds_off(&sB[1][bkr * 128 + bseg]);
    const unsigned dstB1b = lds_off(&sB[1][bkr * 128 + bseg + 8]);

    const int mrow  = wm * 16 + (lane & 15);
    const int khalf = (lane >> 4) * 8;

    v8f acc0 = {};
    v8f acc1 = {};

    #define C2_ISSUE(step)                                                        \
    {                                                                             \
        const int kb = (step) * 32;                                               \
        const _Float16* ga  = act1 + (size_t)arow * O1 + kb + aseg;               \
        const _Float16* gb0 = w2t + (size_t)(kb + bkr) * O2 + n0 + bseg;          \
        const int s1 = (step) & 1;                                                \
        async_copy_b64(s1 ? dstA1 : dstA0, ga);                                   \
        async_copy_b128(s1 ? dstB1a : dstB0a, gb0);                               \
        async_copy_b128(s1 ? dstB1b : dstB0b, gb0 + 8);                           \
    }

    C2_ISSUE(0)
    for (int step = 0; step < K2 / 32; ++step) {
        wait_async0();
        __syncthreads();
        if (step + 1 < K2 / 32) C2_ISSUE(step + 1)
        const int st = step & 1;

        const v8h alo = *(const v8h*)(&sA[st][mrow * 32 + khalf]);
        const v8h ahi = *(const v8h*)(&sA[st][mrow * 32 + 16 + khalf]);
        v16h a;
        #pragma unroll
        for (int i = 0; i < 8; ++i) { a[i] = alo[i]; a[i + 8] = ahi[i]; }
        v16h b0, b1;
        #pragma unroll
        for (int i = 0; i < 8; ++i) {
            b0[i]     = sB[st][(khalf + i) * 128 + (wnb + 0) * 16 + (lane & 15)];
            b0[i + 8] = sB[st][(16 + khalf + i) * 128 + (wnb + 0) * 16 + (lane & 15)];
            b1[i]     = sB[st][(khalf + i) * 128 + (wnb + 1) * 16 + (lane & 15)];
            b1[i + 8] = sB[st][(16 + khalf + i) * 128 + (wnb + 1) * 16 + (lane & 15)];
        }
        acc0 = __builtin_amdgcn_wmma_f32_16x16x32_f16(false, a, false, b0, (short)0, acc0, false, false);
        acc1 = __builtin_amdgcn_wmma_f32_16x16x32_f16(false, a, false, b1, (short)0, acc1, false, false);
    }
    #undef C2_ISSUE

    // epilogue: + bias, raw f32 to workspace (post pass applies validity mask)
    const int mbase = m0 + wm * 16 + ((lane >> 4) ? 8 : 0);
    #pragma unroll
    for (int t = 0; t < 2; ++t) {
        const v8f acc = t ? acc1 : acc0;
        const int ng  = n0 + (wnb + t) * 16 + (lane & 15);
        const float bv = bias[ng];
        #pragma unroll
        for (int r = 0; r < 8; ++r) {
            x2[(size_t)(mbase + r) * O2 + ng] = acc[r] + bv;
        }
    }
}

// ===================================================================
// Post: per (b,p) channel-sum validity, masked writeback, valid counts
// ===================================================================
__global__ __launch_bounds__(256) void k_post(const float* __restrict__ x2,
                                              float*       __restrict__ out,
                                              int*         __restrict__ offs) {
    const int p   = blockIdx.x;            // 0..799 == b*100 + pos
    const int tid = threadIdx.x;
    __shared__ float red[256];
    const float* row = x2 + (size_t)p * O2;
    float s = 0.f;
    #pragma unroll
    for (int j = 0; j < O2 / 256; ++j) s += row[tid + j * 256];
    red[tid] = s;
    __syncthreads();
    for (int st = 128; st > 0; st >>= 1) {
        if (tid < st) red[tid] += red[tid + st];
        __syncthreads();
    }
    const float scale = (red[0] != 0.0f) ? 1.0f : 0.0f;
    #pragma unroll
    for (int j = 0; j < O2 / 256; ++j) {
        const int c = tid + j * 256;
        out[(size_t)p * O2 + c] = row[c] * scale;
    }
    if (tid == 0 && scale != 0.0f) atomicAdd(&offs[p / 100], 1);
}

__global__ void k_finalize(const int* __restrict__ offs, float* __restrict__ out) {
    if (threadIdx.x < 8) out[OUT_MAIN + threadIdx.x] = (float)(offs[threadIdx.x] + 1);
}

// ===================================================================
extern "C" void kernel_launch(void* const* d_in, const int* in_sizes, int n_in,
                              void* d_out, int out_size, void* d_ws, size_t ws_size,
                              hipStream_t stream) {
    const float* voxels   = (const float*)d_in[0];
    const int*   num_pts  = (const int*)d_in[1];
    const int*   coors    = (const int*)d_in[2];
    const float* w_pfn    = (const float*)d_in[3];
    const float* bn_gamma = (const float*)d_in[4];
    const float* bn_beta  = (const float*)d_in[5];
    const float* bn_mean  = (const float*)d_in[6];
    const float* bn_var   = (const float*)d_in[7];
    const float* conv1_w  = (const float*)d_in[8];
    const float* conv1_b  = (const float*)d_in[9];
    const float* conv2_w  = (const float*)d_in[10];
    const float* conv2_b  = (const float*)d_in[11];

    char* ws = (char*)d_ws;
    _Float16* canvas = (_Float16*)(ws + WS_CANVAS);
    _Float16* w1t    = (_Float16*)(ws + WS_W1T);
    _Float16* act1   = (_Float16*)(ws + WS_ACT1);
    _Float16* w2t    = (_Float16*)(ws + WS_W2T);
    float*    x2     = (float*)(ws + WS_X2);
    int*      offs   = (int*)(ws + WS_OFFS);
    float*    zerop  = (float*)(ws + WS_ZERO);
    float*    out    = (float*)d_out;

    k_prep_w1<<<(O1 * K1) / 256, 256, 0, stream>>>(conv1_w, w1t, offs, zerop);
    k_prep_w2<<<(O2 * K2) / 256, 256, 0, stream>>>(conv2_w, w2t);
    k_pfn<<<M_PIL, 256, 0, stream>>>(voxels, num_pts, coors, w_pfn,
                                     bn_gamma, bn_beta, bn_mean, bn_var, canvas);
    k_conv1<<<dim3(M1 / 64, O1 / 64), 256, 0, stream>>>(canvas, w1t, conv1_b,
                                                        (const _Float16*)zerop, act1);
    k_conv2<<<dim3(M2 / 32, O2 / 128), 256, 0, stream>>>(act1, w2t, conv2_b, x2);
    k_post<<<M2, 256, 0, stream>>>(x2, out, offs);
    k_finalize<<<1, 32, 0, stream>>>(offs, out);
}
